// GraphSST2Net_12283606466687
// MI455X (gfx1250) — matrix operand order repaired and checked by reading, not compiled, not run on hardware
//
#include <hip/hip_runtime.h>
#include <hip/hip_bf16.h>

typedef __attribute__((ext_vector_type(16))) __bf16 v16bf;
typedef __attribute__((ext_vector_type(8)))  float  v8f;

#define GD_IN  768
#define GD_H   128
#define GD_MLP 32
#define GD_OUT 2

#if defined(__has_builtin)
#if __has_builtin(__builtin_amdgcn_sched_barrier)
#define SCHED_BARRIER() __builtin_amdgcn_sched_barrier(0)
#endif
#endif
#ifndef SCHED_BARRIER
#define SCHED_BARRIER()
#endif

// ---------------------------------------------------------------------------
// Weight fuse + fp32->bf16 transpose: Wt[n][k] = (n<128 ? Wi : Wr)[k][n%128]
// Wt is [256][K] row-major bf16 so B fragments are two contiguous 16B reads.
// ---------------------------------------------------------------------------
__global__ __launch_bounds__(256) void fuse_w_kernel(
    const float* __restrict__ Wi, const float* __restrict__ Wr,
    __bf16* __restrict__ Wt, int K)
{
  int idx = blockIdx.x * 256 + threadIdx.x;
  if (idx >= 256 * K) return;
  int n = idx / K, k = idx - n * K;
  float v = (n < 128) ? Wi[k * 128 + n] : Wr[k * 128 + (n - 128)];
  Wt[idx] = (__bf16)v;
}

__global__ __launch_bounds__(256) void zero_kernel(float* __restrict__ p, long n)
{
  long i = (long)blockIdx.x * 256 + threadIdx.x;
  if (i < n) p[i] = 0.0f;
}

// ---------------------------------------------------------------------------
// Edge degree: deg[col] += ew   (atomics land in L2; deg is 400KB)
// ---------------------------------------------------------------------------
__global__ __launch_bounds__(256) void deg_kernel(
    const int* __restrict__ col, const float* __restrict__ ew,
    float* __restrict__ deg, int E)
{
  int e = blockIdx.x * 256 + threadIdx.x;
  if (e < E) atomicAdd(&deg[col[e]], ew[e]);
}

// norm[e] = dinv(deg[row]) * ew * dinv(deg[col])
__global__ __launch_bounds__(256) void norm_kernel(
    const int* __restrict__ row, const int* __restrict__ col,
    const float* __restrict__ ew, const float* __restrict__ deg,
    float* __restrict__ norm, int E)
{
  int e = blockIdx.x * 256 + threadIdx.x;
  if (e >= E) return;
  float dr = deg[row[e]], dc = deg[col[e]];
  float ir = dr > 0.0f ? rsqrtf(dr) : 0.0f;
  float ic = dc > 0.0f ? rsqrtf(dc) : 0.0f;
  norm[e] = ir * ew[e] * ic;
}

// ---------------------------------------------------------------------------
// A fragment loader: 8 fp32 at +0 and 8 at +16 (per ISA 16-bit A layout) -> v16bf.
// lane%16 = M-row; lane/16 (khalf) selects which K-halves the lane holds:
//   element j<8  -> K = khalf*8 + j
//   element j>=8 -> K = 16 + khalf*8 + (j-8)
// ---------------------------------------------------------------------------
__device__ __forceinline__ v16bf load_a_frag(const float* __restrict__ ap)
{
  float4 l0 = *(const float4*)(ap);
  float4 l1 = *(const float4*)(ap + 4);
  float4 h0 = *(const float4*)(ap + 16);
  float4 h1 = *(const float4*)(ap + 20);
  v16bf a;
  a[0]  = (__bf16)l0.x; a[1]  = (__bf16)l0.y; a[2]  = (__bf16)l0.z; a[3]  = (__bf16)l0.w;
  a[4]  = (__bf16)l1.x; a[5]  = (__bf16)l1.y; a[6]  = (__bf16)l1.z; a[7]  = (__bf16)l1.w;
  a[8]  = (__bf16)h0.x; a[9]  = (__bf16)h0.y; a[10] = (__bf16)h0.z; a[11] = (__bf16)h0.w;
  a[12] = (__bf16)h1.x; a[13] = (__bf16)h1.y; a[14] = (__bf16)h1.z; a[15] = (__bf16)h1.w;
  return a;
}

// ---------------------------------------------------------------------------
// Dual-output BF16 WMMA GEMM: [nrows x K] fp32 @ (Wt^T) -> outI/outR [nrows x 128]
// Block = 256 thr = 8 waves arranged 2(M) x 4(N); block tile = 64 x 256.
// Wave computes 32 rows x 64 cols: 2 A fragments x 4 B fragments = 8 v8f accs.
// B fragment: lane%16 = N-col, 8 bf16 at K = kb + (lane/16)*8 and 8 at K = kb + 16 + (lane/16)*8.
// sched_barrier splits each K-step into a load/convert phase and a WMMA phase
// so the compiler can't sink individual B loads between WMMAs; with the loop
// fully unrolled it may instead hoist iteration i+1 loads among iteration i
// WMMAs (software pipelining).
// ---------------------------------------------------------------------------
template <int K>
__global__ __launch_bounds__(256) void gemm_bf16_dual(
    const float* __restrict__ X, const __bf16* __restrict__ Wt,
    float* __restrict__ outI, float* __restrict__ outR, int nrows)
{
  const int lane  = threadIdx.x & 31;
  const int wave  = threadIdx.x >> 5;
  const int wm    = wave >> 2;           // 0..1  (M)
  const int wn    = wave & 3;            // 0..3  (N)
  const int row_tile = blockIdx.x * 64 + wm * 32;
  if (row_tile >= nrows) return;         // whole wave exits (N % 32 == 0)
  const int m     = lane & 15;
  const int khalf = lane >> 4;           // 0 or 1
  const float* xrow0 = X + (size_t)(row_tile + m) * K;
  const float* xrow1 = X + (size_t)(row_tile + 16 + m) * K;
  const int nb = wn * 64;
  const __bf16* wbase = Wt + (size_t)(nb + m) * K + khalf * 8;
  auto ldb = [](const __bf16* p) -> v16bf {
    struct { uint4 lo, hi; } t = { *(const uint4*)(p), *(const uint4*)(p + 16) };
    return __builtin_bit_cast(v16bf, t);
  };

  v8f acc0 = {}, acc1 = {}, acc2 = {}, acc3 = {};
  v8f acc4 = {}, acc5 = {}, acc6 = {}, acc7 = {};

#pragma unroll
  for (int kb = 0; kb < K; kb += 32) {
    if (kb + 32 < K) {
      __builtin_prefetch(xrow0 + kb + 64, 0, 1);   // global_prefetch
      __builtin_prefetch(xrow1 + kb + 64, 0, 1);
    }
    // ---- load/convert phase ----
    v16bf a0 = load_a_frag(xrow0 + kb + khalf * 8);
    v16bf a1 = load_a_frag(xrow1 + kb + khalf * 8);
    const __bf16* wp = wbase + kb;
    v16bf b0 = ldb(wp);
    v16bf b1 = ldb(wp + 16 * K);
    v16bf b2 = ldb(wp + 32 * K);
    v16bf b3 = ldb(wp + 48 * K);
    SCHED_BARRIER();
    // ---- WMMA phase: 8 back-to-back matrix ops, B reused for both A frags ----
    acc0 = __builtin_amdgcn_wmma_f32_16x16x32_bf16(false, a0, false, b0, (short)0, acc0, false, false);
    acc1 = __builtin_amdgcn_wmma_f32_16x16x32_bf16(false, a0, false, b1, (short)0, acc1, false, false);
    acc2 = __builtin_amdgcn_wmma_f32_16x16x32_bf16(false, a0, false, b2, (short)0, acc2, false, false);
    acc3 = __builtin_amdgcn_wmma_f32_16x16x32_bf16(false, a0, false, b3, (short)0, acc3, false, false);
    acc4 = __builtin_amdgcn_wmma_f32_16x16x32_bf16(false, a1, false, b0, (short)0, acc4, false, false);
    acc5 = __builtin_amdgcn_wmma_f32_16x16x32_bf16(false, a1, false, b1, (short)0, acc5, false, false);
    acc6 = __builtin_amdgcn_wmma_f32_16x16x32_bf16(false, a1, false, b2, (short)0, acc6, false, false);
    acc7 = __builtin_amdgcn_wmma_f32_16x16x32_bf16(false, a1, false, b3, (short)0, acc7, false, false);
  }

  // ---- store C: lane%16 = N-col, VGPR r = row (r + 8*khalf) ----
  float* outp = (wn < 2) ? outI : outR;
  const int cb = (wn & 1) * 64;
#pragma unroll
  for (int r = 0; r < 8; ++r) {
    const size_t r0 = (size_t)(row_tile + r + 8 * khalf) * GD_H;
    outp[r0 + cb +  0 + m] = acc0[r];
    outp[r0 + cb + 16 + m] = acc1[r];
    outp[r0 + cb + 32 + m] = acc2[r];
    outp[r0 + cb + 48 + m] = acc3[r];
    const size_t r1 = r0 + (size_t)16 * GD_H;
    outp[r1 + cb +  0 + m] = acc4[r];
    outp[r1 + cb + 16 + m] = acc5[r];
    outp[r1 + cb + 32 + m] = acc6[r];
    outp[r1 + cb + 48 + m] = acc7[r];
  }
}

// ---------------------------------------------------------------------------
// Edge scatter: agg[col] += src[row] * norm.  1 wave per edge, float4 per lane.
// agg buffer (51MB) is L2-resident on MI455X (192MB L2).
// ---------------------------------------------------------------------------
__global__ __launch_bounds__(256) void scatter_kernel(
    const int* __restrict__ row, const int* __restrict__ col,
    const float* __restrict__ norm, const float* __restrict__ src,
    float* __restrict__ agg, int E)
{
  int e = blockIdx.x * 8 + (threadIdx.x >> 5);
  if (e >= E) return;
  const int lane = threadIdx.x & 31;
  const int r = row[e], c = col[e];
  const float nv = norm[e];
  const float4 v = *(const float4*)(src + (size_t)r * GD_H + lane * 4);
  float* dst = agg + (size_t)c * GD_H + lane * 4;
  atomicAdd(dst + 0, v.x * nv);
  atomicAdd(dst + 1, v.y * nv);
  atomicAdd(dst + 2, v.z * nv);
  atomicAdd(dst + 3, v.w * nv);
}

// out = relu(agg + lin + bias)   (elu(relu(x)) == relu(x), so layer 1 fuses too)
__global__ __launch_bounds__(256) void act_kernel(
    const float* __restrict__ agg, const float* __restrict__ lin,
    const float* __restrict__ bias, float* __restrict__ out, long total)
{
  long i = (long)blockIdx.x * 256 + threadIdx.x;
  if (i >= total) return;
  float v = agg[i] + lin[i] + bias[(int)(i & (GD_H - 1))];
  out[i] = v > 0.0f ? v : 0.0f;
}

// ---------------------------------------------------------------------------
// Mean-pool accumulate: gsum[batch[i]] += node_x[i]; gcnt[batch[i]] += 1
// ---------------------------------------------------------------------------
__global__ __launch_bounds__(256) void pool_kernel(
    const float* __restrict__ nodex, const int* __restrict__ batch,
    float* __restrict__ gsum, float* __restrict__ gcnt, int N)
{
  int i = blockIdx.x * 8 + (threadIdx.x >> 5);
  if (i >= N) return;
  const int lane = threadIdx.x & 31;
  const int g = batch[i];
  const float4 v = *(const float4*)(nodex + (size_t)i * GD_H + lane * 4);
  float* d = gsum + (size_t)g * GD_H + lane * 4;
  atomicAdd(d + 0, v.x);
  atomicAdd(d + 1, v.y);
  atomicAdd(d + 2, v.z);
  atomicAdd(d + 3, v.w);
  if (lane == 0) atomicAdd(&gcnt[g], 1.0f);
}

// ---------------------------------------------------------------------------
// Head: pred = elu(mean @ Wl1 + bl1) @ Wl2 + bl2.  1 wave per graph (G=2000).
// ---------------------------------------------------------------------------
__global__ __launch_bounds__(256) void head_kernel(
    const float* __restrict__ gsum, const float* __restrict__ gcnt,
    const float* __restrict__ Wl1, const float* __restrict__ bl1,
    const float* __restrict__ Wl2, const float* __restrict__ bl2,
    float* __restrict__ out, int G)
{
  int g = blockIdx.x * 8 + (threadIdx.x >> 5);
  if (g >= G) return;
  const int j = threadIdx.x & 31;                 // lane = MLP unit
  const float inv = 1.0f / fmaxf(gcnt[g], 1.0f);
  const float* gs = gsum + (size_t)g * GD_H;
  float acc = 0.0f;
#pragma unroll 4
  for (int f = 0; f < GD_H; ++f)
    acc = fmaf(gs[f] * inv, Wl1[f * GD_MLP + j], acc);
  acc += bl1[j];
  float z = acc > 0.0f ? acc : (expf(acc) - 1.0f);   // elu
  float p0 = z * Wl2[j * GD_OUT + 0];
  float p1 = z * Wl2[j * GD_OUT + 1];
#pragma unroll
  for (int off = 16; off > 0; off >>= 1) {
    p0 += __shfl_xor(p0, off, 32);
    p1 += __shfl_xor(p1, off, 32);
  }
  if (j == 0) {
    out[g * GD_OUT + 0] = p0 + bl2[0];
    out[g * GD_OUT + 1] = p1 + bl2[1];
  }
}

// ---------------------------------------------------------------------------
extern "C" void kernel_launch(void* const* d_in, const int* in_sizes, int n_in,
                              void* d_out, int out_size, void* d_ws, size_t ws_size,
                              hipStream_t stream)
{
  const float* x    = (const float*)d_in[0];
  const int*   ei   = (const int*)d_in[1];          // [2,E]
  const float* ea   = (const float*)d_in[2];        // [E]
  const int*   bat  = (const int*)d_in[3];          // [N]
  const float* W1i  = (const float*)d_in[5];
  const float* W1r  = (const float*)d_in[6];
  const float* b1   = (const float*)d_in[7];
  const float* W2i  = (const float*)d_in[8];
  const float* W2r  = (const float*)d_in[9];
  const float* b2   = (const float*)d_in[10];
  const float* Wl1  = (const float*)d_in[11];
  const float* bl1  = (const float*)d_in[12];
  const float* Wl2  = (const float*)d_in[13];
  const float* bl2  = (const float*)d_in[14];
  float* out = (float*)d_out;

  const int N = in_sizes[0] / GD_IN;     // 100000
  const int E = in_sizes[2];             // 1600000
  const int G = out_size / GD_OUT;       // 2000
  const int* erow = ei;
  const int* ecol = ei + E;

  // ---- workspace layout (256B aligned slices) ----
  char* ws = (char*)d_ws;
  size_t off = 0;
  auto alloc = [&](size_t bytes) -> void* {
    void* p = ws + off;
    off = (off + bytes + 255) & ~(size_t)255;
    return p;
  };
  __bf16* Wt1  = (__bf16*)alloc((size_t)256 * GD_IN * sizeof(__bf16));
  __bf16* Wt2  = (__bf16*)alloc((size_t)256 * GD_H  * sizeof(__bf16));
  float*  deg  = (float*)alloc((size_t)N * sizeof(float));
  float*  norm = (float*)alloc((size_t)E * sizeof(float));
  float*  bufA = (float*)alloc((size_t)N * GD_H * sizeof(float));
  float*  bufB = (float*)alloc((size_t)N * GD_H * sizeof(float));
  float*  bufC = (float*)alloc((size_t)N * GD_H * sizeof(float));
  float*  gsum = (float*)alloc((size_t)G * GD_H * sizeof(float));
  float*  gcnt = (float*)alloc((size_t)G * sizeof(float));
  (void)ws_size; (void)n_in;

  const long NF = (long)N * GD_H;
  const int zb_nf  = (int)((NF + 255) / 256);
  const int eb     = (E + 255) / 256;
  const int eb8    = (E + 7) / 8;
  const int nb8    = (N + 7) / 8;
  const int gemm_b = (N + 63) / 64;

  // 1) weights -> fused bf16 transposed
  fuse_w_kernel<<<(256 * GD_IN + 255) / 256, 256, 0, stream>>>(W1i, W1r, Wt1, GD_IN);
  fuse_w_kernel<<<(256 * GD_H  + 255) / 256, 256, 0, stream>>>(W2i, W2r, Wt2, GD_H);

  // 2) degrees + edge norms
  zero_kernel<<<(N + 255) / 256, 256, 0, stream>>>(deg, N);
  deg_kernel<<<eb, 256, 0, stream>>>(ecol, ea, deg, E);
  norm_kernel<<<eb, 256, 0, stream>>>(erow, ecol, ea, deg, norm, E);

  // 3) layer 1: bufA = x@W1i, bufB = x@W1r   (one pass over 307MB of x)
  gemm_bf16_dual<GD_IN><<<gemm_b, 256, 0, stream>>>(x, Wt1, bufA, bufB, N);
  zero_kernel<<<zb_nf, 256, 0, stream>>>(bufC, NF);
  scatter_kernel<<<eb8, 256, 0, stream>>>(erow, ecol, norm, bufA, bufC, E);
  // h = elu(relu(agg + xW1r + b1)) == relu(...)  -> bufA
  act_kernel<<<zb_nf, 256, 0, stream>>>(bufC, bufB, b1, bufA, NF);

  // 4) layer 2: bufB = h@W2i, bufC = h@W2r
  gemm_bf16_dual<GD_H><<<gemm_b, 256, 0, stream>>>(bufA, Wt2, bufB, bufC, N);
  zero_kernel<<<zb_nf, 256, 0, stream>>>(bufA, NF);
  scatter_kernel<<<eb8, 256, 0, stream>>>(erow, ecol, norm, bufB, bufA, E);
  // node_x = relu(agg + hW2r + b2) -> bufB
  act_kernel<<<zb_nf, 256, 0, stream>>>(bufA, bufC, b2, bufB, NF);

  // 5) mean pool + head
  zero_kernel<<<((long)G * GD_H + 255) / 256, 256, 0, stream>>>(gsum, (long)G * GD_H);
  zero_kernel<<<(G + 255) / 256, 256, 0, stream>>>(gcnt, G);
  pool_kernel<<<nb8, 256, 0, stream>>>(bufB, bat, gsum, gcnt, N);
  head_kernel<<<(G + 7) / 8, 256, 0, stream>>>(gsum, gcnt, Wl1, bl1, Wl2, bl2, out, G);
}
